// DeepSeekMoE_50843822850504
// MI455X (gfx1250) — matrix operand order repaired and checked by reading, not compiled, not run on hardware
//
#include <hip/hip_runtime.h>
#include <hip/hip_bf16.h>
#include <math.h>

// ---------------- problem constants ----------------
#define NTOK   8192      // B*S
#define D_DIM  2048
#define H_DIM  1024
#define E_NUM  8
#define SH_DIM 2048
#define TOPK   2
#define NPAIR  (NTOK * TOPK)          // 16384
#define MAX_TILES 520                 // sum ceil(c_e/32) <= NPAIR/32 + E
#define CAP    16640                  // padded pair capacity (16632 rounded up)

#define A_TILE (32 * 32)              // ushorts per A buffer
#define B_TILE (8 * 512)              // ushorts per B buffer

typedef __bf16 v16bf __attribute__((ext_vector_type(16)));
typedef float  v8f   __attribute__((ext_vector_type(8)));

// ---------------- workspace layout (bytes) ----------------
#define WS_CNT      0            // int[8]
#define WS_OFFP     64           // int[9]
#define WS_CURSOR   128          // int[8]
#define WS_NTILES   192          // int
#define WS_TOPIDX   256                            // int[NPAIR]
#define WS_TOPGATE  (WS_TOPIDX  + NPAIR*4)         // float[NPAIR]
#define WS_TILE_E   (WS_TOPGATE + NPAIR*4)         // int[MAX_TILES]
#define WS_TILE_R   (WS_TILE_E  + MAX_TILES*4)
#define WS_STOK     (WS_TILE_R  + MAX_TILES*4)     // int[CAP]
#define WS_SGATE    (WS_STOK    + CAP*4)           // float[CAP]
#define WS_HROUT    ((WS_SGATE  + CAP*4 + 255) & ~(size_t)255)  // ushort[CAP*H]
#define WS_HSHARED  (WS_HROUT   + (size_t)CAP*H_DIM*2)          // ushort[NTOK*SH]
#define WS_NEEDED   (WS_HSHARED + (size_t)NTOK*SH_DIM*2)

// ---------------- helpers ----------------
__device__ __forceinline__ unsigned short f2bf(float f) {
  union { __bf16 h; unsigned short u; } c;
  c.h = (__bf16)f;
  return c.u;
}

// true packed f32x2 -> bf16x2 convert (single VALU op); clang won't fuse the
// union form, so emit it directly.
__device__ __forceinline__ unsigned pack2bf(float a, float b) {
  unsigned r;
  asm("v_cvt_pk_bf16_f32 %0, %1, %2" : "=v"(r) : "v"(a), "v"(b));
  return r;
}

__device__ __forceinline__ float gelu_erf(float v) {
  return 0.5f * v * (1.0f + erff(v * 0.70710678118654752f));
}

__device__ __forceinline__ v8f wmma_bf16(v16bf a, v16bf b, v8f c) {
  return __builtin_amdgcn_wmma_f32_16x16x32_bf16(false, a, false, b, (short)0, c,
                                                 false, false);
}

// A fragment: 16x32 bf16 (rows rbase..rbase+15 of sA).  ISA layout: lanes0-15 row
// M=lane with K0-7 & K16-23; lanes16-31 row M=lane-16 with K8-15 & K24-31.
__device__ __forceinline__ v16bf ldsA_frag(const unsigned short* sA, int lane) {
  int m  = lane & 15;
  int hi = (lane >> 4) & 1;
  const unsigned short* base = sA + m * 32 + hi * 8;
  union { v16bf v; uint4 q[2]; } u;
  u.q[0] = *(const uint4*)(base);        // K 0..7   (+8 for hi lanes)
  u.q[1] = *(const uint4*)(base + 16);   // K 16..23 (+8 for hi lanes)
  return u.v;
}

// B fragment: 32x16 bf16 per 16-col slice. sB layout: slice*512 + col*32 + k.
// lanes0-15: col=lane, K0-7 & K16-23; lanes16-31: col=lane-16, K8-15 & K24-31.
__device__ __forceinline__ v16bf ldsB_frag(const unsigned short* sB, int lane, int slice) {
  int j  = lane & 15;
  int hi = (lane >> 4) & 1;
  const unsigned short* base = sB + slice * 512 + j * 32 + hi * 8;
  union { v16bf v; uint4 q[2]; } u;
  u.q[0] = *(const uint4*)(base);
  u.q[1] = *(const uint4*)(base + 16);
  return u.v;
}

// ---------------- GEMM core: 32 x 128 tile, K-step 32, double-buffered ----------------
// 4 waves; each wave: 2 A-frags x 2 B-slices = 4 WMMA per K-step.
// AMODE 0: A = f32 rows at aRow0        (shared expert layer 1)
// AMODE 1: A = bf16 rows at aRow0       (layer 2, reads h buffers)
// AMODE 2: A = f32 rows gathered by tok (routed layer 1)
template <int AMODE>
__device__ __forceinline__ void gemm_core(const void* Asrc, int lda, int aRow0,
                                          const int* gatherTok,
                                          const float* Bsrc, int ldb, int n0, int Ksize,
                                          unsigned short* sA, unsigned short* sB,
                                          int tid, int lane, int wave,
                                          v8f acc[4]) {
  int cg = tid & 31;    // B: column group, cols cg*4 .. cg*4+3
  int kr = tid >> 5;    // B: k subrange,  kk = kr*8 .. kr*8+7

  // hoisted per-thread A addressing (token gather resolved once)
  int r_[2], kq_[2];
  size_t abase[2];
  bool valid[2];
  #pragma unroll
  for (int i = 0; i < 2; ++i) {
    int idx = i * 128 + tid;
    r_[i] = idx >> 3;
    kq_[i] = (idx & 7) * 4;
    int row;
    if (AMODE == 2) {
      int t = gatherTok[r_[i]];
      valid[i] = (t >= 0);
      row = valid[i] ? t : 0;
    } else {
      valid[i] = true;
      row = aRow0 + r_[i];
    }
    abase[i] = (size_t)row * lda + kq_[i];
  }
  const float* bbase = Bsrc + (size_t)(kr * 8) * ldb + n0 + cg * 4;

  float4 fa[2];
  uint2  ra[2];
  float4 fb[8];

  auto load_tile = [&](int k0) {
    #pragma unroll
    for (int i = 0; i < 2; ++i) {
      if (AMODE == 1)
        ra[i] = *(const uint2*)((const unsigned short*)Asrc + abase[i] + k0);
      else
        fa[i] = *(const float4*)((const float*)Asrc + abase[i] + k0);
    }
    #pragma unroll
    for (int i = 0; i < 8; ++i)
      fb[i] = *(const float4*)(bbase + (size_t)(k0 + i) * ldb);
  };

  auto store_tile = [&](unsigned short* dA, unsigned short* dB) {
    #pragma unroll
    for (int i = 0; i < 2; ++i) {
      uint2 d;
      if (AMODE == 1) {
        d = ra[i];
      } else {
        d.x = valid[i] ? pack2bf(fa[i].x, fa[i].y) : 0u;
        d.y = valid[i] ? pack2bf(fa[i].z, fa[i].w) : 0u;
      }
      *(uint2*)&dA[r_[i] * 32 + kq_[i]] = d;
    }
    #pragma unroll
    for (int j = 0; j < 4; ++j) {
      int col = cg * 4 + j;
      uint4 d;
      d.x = pack2bf(((const float*)&fb[0])[j], ((const float*)&fb[1])[j]);
      d.y = pack2bf(((const float*)&fb[2])[j], ((const float*)&fb[3])[j]);
      d.z = pack2bf(((const float*)&fb[4])[j], ((const float*)&fb[5])[j]);
      d.w = pack2bf(((const float*)&fb[6])[j], ((const float*)&fb[7])[j]);
      *(uint4*)&dB[(col >> 4) * 512 + (col & 15) * 32 + kr * 8] = d;
    }
  };

  load_tile(0);
  int buf = 0;
  for (int k0 = 0; k0 < Ksize; k0 += 32) {
    unsigned short* dA = sA + buf * A_TILE;
    unsigned short* dB = sB + buf * B_TILE;
    store_tile(dA, dB);
    if (k0 + 32 < Ksize) load_tile(k0 + 32);   // overlap VMEM with barrier+WMMA
    __syncthreads();                           // single barrier per step (2 buffers)
    v16bf a0 = ldsA_frag(dA, lane);
    v16bf a1 = ldsA_frag(dA + 16 * 32, lane);
    v16bf b0 = ldsB_frag(dB, lane, wave * 2);
    v16bf b1 = ldsB_frag(dB, lane, wave * 2 + 1);
    acc[0] = wmma_bf16(a0, b0, acc[0]);
    acc[1] = wmma_bf16(a0, b1, acc[1]);
    acc[2] = wmma_bf16(a1, b0, acc[2]);
    acc[3] = wmma_bf16(a1, b1, acc[3]);
    buf ^= 1;
  }
}

// ---------------- kernels ----------------
__global__ __launch_bounds__(256)
void init_kernel(int* cnt, int* sorted_tok, float* sorted_gate) {
  int i = blockIdx.x * blockDim.x + threadIdx.x;
  if (i < E_NUM) cnt[i] = 0;
  if (i < CAP) { sorted_tok[i] = -1; sorted_gate[i] = 0.0f; }
}

__global__ __launch_bounds__(256)
void gate_kernel(const float* __restrict__ x, const float* __restrict__ gw,
                 int* top_idx, float* top_gate, int* cnt) {
  int wid  = blockIdx.x * (blockDim.x >> 5) + (threadIdx.x >> 5);
  int lane = threadIdx.x & 31;
  if (wid >= NTOK) return;
  float acc[E_NUM];
  #pragma unroll
  for (int e = 0; e < E_NUM; ++e) acc[e] = 0.0f;
  const float* xr = x + (size_t)wid * D_DIM;
  for (int d = lane; d < D_DIM; d += 32) {
    float xv = xr[d];
    const float4 g0 = *(const float4*)(gw + (size_t)d * E_NUM);
    const float4 g1 = *(const float4*)(gw + (size_t)d * E_NUM + 4);
    acc[0] += xv * g0.x; acc[1] += xv * g0.y; acc[2] += xv * g0.z; acc[3] += xv * g0.w;
    acc[4] += xv * g1.x; acc[5] += xv * g1.y; acc[6] += xv * g1.z; acc[7] += xv * g1.w;
  }
  #pragma unroll
  for (int off = 16; off > 0; off >>= 1) {
    #pragma unroll
    for (int e = 0; e < E_NUM; ++e) acc[e] += __shfl_xor(acc[e], off, 32);
  }
  if (lane == 0) {
    int b0 = 0; float l0 = acc[0];
    #pragma unroll
    for (int e = 1; e < E_NUM; ++e) if (acc[e] > l0) { l0 = acc[e]; b0 = e; }
    int b1 = -1; float l1 = -3.4e38f;
    #pragma unroll
    for (int e = 0; e < E_NUM; ++e)
      if (e != b0 && acc[e] > l1) { l1 = acc[e]; b1 = e; }
    float ex = __expf(l1 - l0);
    float inv = 1.0f / (1.0f + ex);
    top_idx[wid * 2]     = b0;  top_idx[wid * 2 + 1]  = b1;
    top_gate[wid * 2]    = inv; top_gate[wid * 2 + 1] = ex * inv;
    atomicAdd(&cnt[b0], 1);
    atomicAdd(&cnt[b1], 1);
  }
}

__global__ void scan_kernel(const int* cnt, int* offp, int* cursor,
                            int* tile_e, int* tile_r, int* ntiles) {
  if (threadIdx.x != 0 || blockIdx.x != 0) return;
  int off = 0, t = 0;
  for (int e = 0; e < E_NUM; ++e) {
    cursor[e] = 0;
    offp[e] = off;
    int nt = (cnt[e] + 31) >> 5;              // 32-row tiles
    for (int i = 0; i < nt; ++i) { tile_e[t] = e; tile_r[t] = off + i * 32; ++t; }
    off += nt * 32;
  }
  offp[E_NUM] = off;
  *ntiles = t;
}

__global__ __launch_bounds__(256)
void scatter_kernel(const int* top_idx, const float* top_gate, const int* offp,
                    int* cursor, int* sorted_tok, float* sorted_gate) {
  int p = blockIdx.x * blockDim.x + threadIdx.x;
  if (p >= NPAIR) return;
  int e = top_idx[p];
  int r = atomicAdd(&cursor[e], 1);
  int pos = offp[e] + r;
  sorted_tok[pos]  = p >> 1;
  sorted_gate[pos] = top_gate[p];
}

// shared expert layer 1: h_shared = gelu(x @ sw1 + sb1), bf16 out
__global__ __launch_bounds__(128)
void gemm1_shared_kernel(const float* __restrict__ x, const float* __restrict__ w,
                         const float* __restrict__ bias, unsigned short* __restrict__ h) {
  __shared__ unsigned short sA[2 * A_TILE];
  __shared__ unsigned short sB[2 * B_TILE];
  int tid = threadIdx.x, lane = tid & 31, wave = tid >> 5;
  int m0 = blockIdx.x * 32, n0 = blockIdx.y * 128;
  v8f acc[4] = {};
  gemm_core<0>(x, D_DIM, m0, nullptr, w, SH_DIM, n0, D_DIM, sA, sB, tid, lane, wave, acc);
  int col = lane & 15, hb = lane >> 4;
  #pragma unroll
  for (int v = 0; v < 8; ++v) {
    int rlo = m0 + v + (hb << 3), rhi = rlo + 16;
    int na = n0 + wave * 32 + col, nb = na + 16;
    float ba = bias[na], bb = bias[nb];
    h[(size_t)rlo * SH_DIM + na] = f2bf(gelu_erf(acc[0][v] + ba));
    h[(size_t)rlo * SH_DIM + nb] = f2bf(gelu_erf(acc[1][v] + bb));
    h[(size_t)rhi * SH_DIM + na] = f2bf(gelu_erf(acc[2][v] + ba));
    h[(size_t)rhi * SH_DIM + nb] = f2bf(gelu_erf(acc[3][v] + bb));
  }
}

// shared expert layer 2: out = h_shared @ sw2 + sb2 (plain store, defines out)
__global__ __launch_bounds__(128)
void gemm2_shared_kernel(const unsigned short* __restrict__ h,
                         const float* __restrict__ w, const float* __restrict__ bias,
                         float* __restrict__ out) {
  __shared__ unsigned short sA[2 * A_TILE];
  __shared__ unsigned short sB[2 * B_TILE];
  int tid = threadIdx.x, lane = tid & 31, wave = tid >> 5;
  int m0 = blockIdx.x * 32, n0 = blockIdx.y * 128;
  v8f acc[4] = {};
  gemm_core<1>(h, SH_DIM, m0, nullptr, w, D_DIM, n0, SH_DIM, sA, sB, tid, lane, wave, acc);
  int col = lane & 15, hb = lane >> 4;
  #pragma unroll
  for (int v = 0; v < 8; ++v) {
    int rlo = m0 + v + (hb << 3), rhi = rlo + 16;
    int na = n0 + wave * 32 + col, nb = na + 16;
    float ba = bias[na], bb = bias[nb];
    out[(size_t)rlo * D_DIM + na] = acc[0][v] + ba;
    out[(size_t)rlo * D_DIM + nb] = acc[1][v] + bb;
    out[(size_t)rhi * D_DIM + na] = acc[2][v] + ba;
    out[(size_t)rhi * D_DIM + nb] = acc[3][v] + bb;
  }
}

// routed layer 1: per-tile (expert e), h_routed[pos] = gelu(x[tok] @ w1[e] + b1[e])
__global__ __launch_bounds__(128)
void gemm1_routed_kernel(const float* __restrict__ x, const float* __restrict__ w1,
                         const float* __restrict__ b1, const int* __restrict__ sorted_tok,
                         const int* __restrict__ tile_e, const int* __restrict__ tile_r,
                         const int* __restrict__ ntiles, unsigned short* __restrict__ h) {
  if ((int)blockIdx.x >= *ntiles) return;
  __shared__ unsigned short sA[2 * A_TILE];
  __shared__ unsigned short sB[2 * B_TILE];
  int tid = threadIdx.x, lane = tid & 31, wave = tid >> 5;
  int e = tile_e[blockIdx.x], row0 = tile_r[blockIdx.x];
  int n0 = blockIdx.y * 128;
  const float* w  = w1 + (size_t)e * D_DIM * H_DIM;
  const float* be = b1 + (size_t)e * H_DIM;
  v8f acc[4] = {};
  gemm_core<2>(x, D_DIM, 0, sorted_tok + row0, w, H_DIM, n0, D_DIM, sA, sB, tid, lane,
               wave, acc);
  int col = lane & 15, hb = lane >> 4;
  #pragma unroll
  for (int v = 0; v < 8; ++v) {
    int rlo = row0 + v + (hb << 3), rhi = rlo + 16;
    int na = n0 + wave * 32 + col, nb = na + 16;
    float ba = be[na], bb = be[nb];
    h[(size_t)rlo * H_DIM + na] = f2bf(gelu_erf(acc[0][v] + ba));
    h[(size_t)rlo * H_DIM + nb] = f2bf(gelu_erf(acc[1][v] + bb));
    h[(size_t)rhi * H_DIM + na] = f2bf(gelu_erf(acc[2][v] + ba));
    h[(size_t)rhi * H_DIM + nb] = f2bf(gelu_erf(acc[3][v] + bb));
  }
}

// routed layer 2: out[tok] += gate * (h_routed[pos] @ w2[e] + b2[e])
__global__ __launch_bounds__(128)
void gemm2_routed_kernel(const unsigned short* __restrict__ h,
                         const float* __restrict__ w2, const float* __restrict__ b2,
                         const int* __restrict__ sorted_tok,
                         const float* __restrict__ sorted_gate,
                         const int* __restrict__ tile_e, const int* __restrict__ tile_r,
                         const int* __restrict__ ntiles, float* __restrict__ out) {
  if ((int)blockIdx.x >= *ntiles) return;
  __shared__ unsigned short sA[2 * A_TILE];
  __shared__ unsigned short sB[2 * B_TILE];
  int tid = threadIdx.x, lane = tid & 31, wave = tid >> 5;
  int e = tile_e[blockIdx.x], row0 = tile_r[blockIdx.x];
  int n0 = blockIdx.y * 128;
  const float* w  = w2 + (size_t)e * H_DIM * D_DIM;
  const float* be = b2 + (size_t)e * D_DIM;
  v8f acc[4] = {};
  gemm_core<1>(h, H_DIM, row0, nullptr, w, D_DIM, n0, H_DIM, sA, sB, tid, lane, wave, acc);
  int col = lane & 15, hb = lane >> 4;
  #pragma unroll
  for (int v = 0; v < 8; ++v) {
    int rlo = row0 + v + (hb << 3), rhi = rlo + 16;
    int na = n0 + wave * 32 + col, nb = na + 16;
    float ba = be[na], bb = be[nb];
    int tlo = sorted_tok[rlo], thi = sorted_tok[rhi];
    if (tlo >= 0) {
      float g = sorted_gate[rlo];
      atomicAdd(&out[(size_t)tlo * D_DIM + na], g * (acc[0][v] + ba));
      atomicAdd(&out[(size_t)tlo * D_DIM + nb], g * (acc[1][v] + bb));
    }
    if (thi >= 0) {
      float g = sorted_gate[rhi];
      atomicAdd(&out[(size_t)thi * D_DIM + na], g * (acc[2][v] + ba));
      atomicAdd(&out[(size_t)thi * D_DIM + nb], g * (acc[3][v] + bb));
    }
  }
}

// ---------------- launcher ----------------
extern "C" void kernel_launch(void* const* d_in, const int* in_sizes, int n_in,
                              void* d_out, int out_size, void* d_ws, size_t ws_size,
                              hipStream_t stream) {
  (void)in_sizes; (void)n_in; (void)out_size;
  if (ws_size < WS_NEEDED) return;

  const float* x   = (const float*)d_in[0];
  const float* gw  = (const float*)d_in[1];
  const float* w1  = (const float*)d_in[2];
  const float* b1  = (const float*)d_in[3];
  const float* w2  = (const float*)d_in[4];
  const float* b2  = (const float*)d_in[5];
  const float* sw1 = (const float*)d_in[6];
  const float* sb1 = (const float*)d_in[7];
  const float* sw2 = (const float*)d_in[8];
  const float* sb2 = (const float*)d_in[9];
  float* out = (float*)d_out;

  char* ws = (char*)d_ws;
  int*   cnt      = (int*)(ws + WS_CNT);
  int*   offp     = (int*)(ws + WS_OFFP);
  int*   cursor   = (int*)(ws + WS_CURSOR);
  int*   ntiles   = (int*)(ws + WS_NTILES);
  int*   top_idx  = (int*)(ws + WS_TOPIDX);
  float* top_gate = (float*)(ws + WS_TOPGATE);
  int*   tile_e   = (int*)(ws + WS_TILE_E);
  int*   tile_r   = (int*)(ws + WS_TILE_R);
  int*   s_tok    = (int*)(ws + WS_STOK);
  float* s_gate   = (float*)(ws + WS_SGATE);
  unsigned short* h_rout = (unsigned short*)(ws + WS_HROUT);
  unsigned short* h_shar = (unsigned short*)(ws + WS_HSHARED);

  init_kernel<<<(CAP + 255) / 256, 256, 0, stream>>>(cnt, s_tok, s_gate);
  gate_kernel<<<NTOK / 8, 256, 0, stream>>>(x, gw, top_idx, top_gate, cnt);
  scan_kernel<<<1, 1, 0, stream>>>(cnt, offp, cursor, tile_e, tile_r, ntiles);
  scatter_kernel<<<NPAIR / 256, 256, 0, stream>>>(top_idx, top_gate, offp, cursor,
                                                  s_tok, s_gate);
  // shared experts: writes full `out`
  gemm1_shared_kernel<<<dim3(NTOK / 32, SH_DIM / 128), 128, 0, stream>>>(x, sw1, sb1,
                                                                         h_shar);
  gemm2_shared_kernel<<<dim3(NTOK / 32, D_DIM / 128), 128, 0, stream>>>(h_shar, sw2,
                                                                        sb2, out);
  // routed experts: grouped GEMM over sorted pairs, atomic combine into `out`
  gemm1_routed_kernel<<<dim3(MAX_TILES, H_DIM / 128), 128, 0, stream>>>(
      x, w1, b1, s_tok, tile_e, tile_r, ntiles, h_rout);
  gemm2_routed_kernel<<<dim3(MAX_TILES, D_DIM / 128), 128, 0, stream>>>(
      h_rout, w2, b2, s_tok, s_gate, tile_e, tile_r, ntiles, out);
}
